// TMix_gptalpha_7524782702652
// MI455X (gfx1250) — matrix operand order repaired and checked by reading, not compiled, not run on hardware
//
#include <hip/hip_runtime.h>
#include <hip/hip_bf16.h>

typedef __attribute__((ext_vector_type(16))) _Float16 v16h;
typedef __attribute__((ext_vector_type(8)))  _Float16 v8h;
typedef __attribute__((ext_vector_type(8)))  float    v8f;

#define CD   1024
#define TD   2048
#define BD   2
#define HD   16
#define KD   64
#define BT   (BD*TD)

// ---------------- WMMA helpers (gfx1250, wave32) ----------------

__device__ __forceinline__ v8f wmma16(v16h a, v16h b, v8f c) {
  // v_wmma_f32_16x16x32_f16 : D = A(16x32 f16) * B(32x16 f16) + C(16x16 f32)
  return __builtin_amdgcn_wmma_f32_16x16x32_f16(false, a, false, b, (short)0, c, false, false);
}

// A-matrix 16x32 f16 fragment. Lane (l&15)=M row. Halves 0..7 <- K = kb+0..7,
// halves 8..15 <- K = kb+16..23, where kb = (l>>4)*8. Caller passes pointer to
// rowBase + kb; we grab [0..7] and [16..23].
__device__ __forceinline__ v16h load_fragA(const _Float16* p) {
  union { v16h v; v8h h[2]; } u;
  u.h[0] = *(const v8h*)(p);
  u.h[1] = *(const v8h*)(p + 16);
  return u.v;
}

// B-matrix 32x16 f16 fragment. Lane (l&15)=N col. Halves 0..7 <- K = kb+0..7,
// halves 8..15 <- K = kb+16..23, where kb = (l>>4)*8. Caller passes pointer to
// colBase + (l>>4)*8.
__device__ __forceinline__ v16h load_fragB(const _Float16* p) {
  union { v16h v; v8h h[2]; } u;
  u.h[0] = *(const v8h*)(p);
  u.h[1] = *(const v8h*)(p + 16);
  return u.v;
}

__device__ __forceinline__ float rmax16(float x) {
  #pragma unroll
  for (int o = 8; o >= 1; o >>= 1) x = fmaxf(x, __shfl_xor(x, o, 16));
  return x;
}
__device__ __forceinline__ float rsum16(float x) {
  #pragma unroll
  for (int o = 8; o >= 1; o >>= 1) x += __shfl_xor(x, o, 16);
  return x;
}

// ---------------- 1) token-shift + tanh-LoRA mixing ----------------
__global__ void mix_kernel(const float* __restrict__ x, const float* __restrict__ sstate,
                           const float* __restrict__ tmx, const float* __restrict__ tmr,
                           const float* __restrict__ tmk, const float* __restrict__ tmv,
                           const float* __restrict__ w1,  const float* __restrict__ w2,
                           _Float16* __restrict__ xq, _Float16* __restrict__ xk,
                           _Float16* __restrict__ xv) {
  __shared__ float sx[CD], sdx[CD], sxx[CD], slora[96];
  int row = blockIdx.x;
  int b = row / TD, t = row % TD;
  const float* xr = x + (size_t)row * CD;
  const float* xp = (t == 0) ? (sstate + (size_t)b * CD) : (xr - CD);
  for (int i = threadIdx.x; i < CD; i += 256) {
    float xc = xr[i];
    float dx = xp[i] - xc;
    sx[i] = xc; sdx[i] = dx;
    sxx[i] = xc + dx * tmx[i];
  }
  __syncthreads();
  if (threadIdx.x < 96) {
    int d = threadIdx.x;
    float acc = 0.f;
    for (int c = 0; c < CD; ++c) acc += sxx[c] * w1[c * 96 + d];
    slora[d] = tanhf(acc);
  }
  __syncthreads();
  for (int i = threadIdx.x; i < CD; i += 256) {
    float m[3];
    #pragma unroll
    for (int l = 0; l < 3; ++l) {
      float a = 0.f;
      for (int d = 0; d < 32; ++d) a += slora[l * 32 + d] * w2[(size_t)(l * 32 + d) * CD + i];
      m[l] = a;
    }
    float xd = sdx[i], xb = sx[i];
    size_t o = (size_t)row * CD + i;
    xq[o] = (_Float16)(xb + xd * (tmr[i] + m[0]));
    xk[o] = (_Float16)(xb + xd * (tmk[i] + m[1]));
    xv[o] = (_Float16)(xb + xd * (tmv[i] + m[2]));
  }
}

// ---------------- 2) fp32 -> f16 convert ----------------
__global__ void cvt_f32_f16(const float* __restrict__ s, _Float16* __restrict__ d, int n) {
  int i = blockIdx.x * blockDim.x + threadIdx.x;
  if (i < n) d[i] = (_Float16)s[i];
}

// ---------------- 3) GEMM  Y[R,1024] = A[R,1024] * W[1024,1024]^T ----------------
// one wave per block; wave computes a 16x64 tile; double-buffered K-loop so
// fragment loads for the next K-group overlap the WMMAs of the current group.
// Register budget: acc 32 + two 5-frag groups 80 + addressing ~= 120 VGPRs
// (spill-free under wave32).
struct FragGroup { v16h a0, b0, b1, b2, b3; };

__device__ __forceinline__ void load_group(FragGroup& f, const _Float16* aB0,
                                           const _Float16* bB, int k0) {
  f.a0 = load_fragA(aB0 + k0);
  f.b0 = load_fragB(bB + k0);
  f.b1 = load_fragB(bB + 16 * CD + k0);
  f.b2 = load_fragB(bB + 32 * CD + k0);
  f.b3 = load_fragB(bB + 48 * CD + k0);
}

__device__ __forceinline__ void mma_group(const FragGroup& f, v8f acc[4]) {
  acc[0] = wmma16(f.a0, f.b0, acc[0]);
  acc[1] = wmma16(f.a0, f.b1, acc[1]);
  acc[2] = wmma16(f.a0, f.b2, acc[2]);
  acc[3] = wmma16(f.a0, f.b3, acc[3]);
}

__global__ void __launch_bounds__(32, 1)
gemm_nt(const _Float16* __restrict__ A, const _Float16* __restrict__ W,
        float* __restrict__ Y) {
  int lane = threadIdx.x;
  int m0 = blockIdx.y * 16;
  int n0 = blockIdx.x * 64;
  int nlo = lane & 15;
  int kA = (lane >> 4) * 8;
  int kB = (lane >> 4) * 8;
  const _Float16* aB0 = A + (size_t)(m0 + nlo) * CD + kA;
  const _Float16* bB  = W + (size_t)(n0 + nlo) * CD + kB;

  v8f acc[4];
  #pragma unroll
  for (int i = 0; i < 4; ++i) acc[i] = (v8f){};

  FragGroup fA, fB;
  load_group(fA, aB0, bB, 0);
  for (int k0 = 0; k0 < CD; k0 += 64) {
    load_group(fB, aB0, bB, k0 + 32);          // in flight during fA math
    mma_group(fA, acc);
    if (k0 + 64 < CD) load_group(fA, aB0, bB, k0 + 64);
    mma_group(fB, acc);
  }

  int mb = (lane >> 4) * 8;
  #pragma unroll
  for (int v = 0; v < 8; ++v) {
    size_t r = (size_t)(m0 + mb + v) * CD + n0 + nlo;
    Y[r]      = acc[0][v];
    Y[r + 16] = acc[1][v];
    Y[r + 32] = acc[2][v];
    Y[r + 48] = acc[3][v];
  }
}

// ---------------- 4) LayerNorm (+RoPE / layout) ----------------
// mode 0: rope, write head-major [BH,T,64] f16  (q,k)
// mode 1: no rope, write transposed [BH,64,T] f16  (v)
// mode 2: no rope, write flat [row,1024] f16  (output LN)
__global__ void ln_kernel(const float* __restrict__ src, const float* __restrict__ g,
                          const float* __restrict__ bb, const float* __restrict__ ang,
                          _Float16* __restrict__ dst, int mode) {
  __shared__ float srow[CD];
  __shared__ float sred[256];
  int row = blockIdx.x, tid = threadIdx.x;
  const float* sp = src + (size_t)row * CD;
  float ls = 0.f;
  for (int i = tid; i < CD; i += 256) { float v = sp[i]; srow[i] = v; ls += v; }
  sred[tid] = ls; __syncthreads();
  for (int o = 128; o > 0; o >>= 1) { if (tid < o) sred[tid] += sred[tid + o]; __syncthreads(); }
  float mean = sred[0] * (1.0f / CD);
  __syncthreads();
  float lv = 0.f;
  for (int i = tid; i < CD; i += 256) { float d = srow[i] - mean; lv += d * d; }
  sred[tid] = lv; __syncthreads();
  for (int o = 128; o > 0; o >>= 1) { if (tid < o) sred[tid] += sred[tid + o]; __syncthreads(); }
  float rstd = rsqrtf(sred[0] * (1.0f / CD) + 1e-5f);
  int t = row % TD, b = row / TD;
  for (int i = tid; i < CD; i += 256) {
    float yn = (srow[i] - mean) * rstd * g[i] + bb[i];
    if (mode == 2) { dst[(size_t)row * CD + i] = (_Float16)yn; continue; }
    int h = i >> 6, idx = i & 63;
    size_t bh = (size_t)b * HD + h;
    if (mode == 1) { dst[(bh * KD + idx) * TD + t] = (_Float16)yn; continue; }
    int half = idx & 31;
    float a = ang[t * 32 + half];
    float cs = cosf(a), sn = sinf(a);
    int pc = (idx < 32) ? (i + 32) : (i - 32);
    float yp = (srow[pc] - mean) * rstd * g[pc] + bb[pc];
    float out = (idx < 32) ? (yn * cs - yp * sn) : (yp * sn + yn * cs);
    dst[(bh * TD + t) * KD + idx] = (_Float16)out;
  }
}

// ---------------- 5) causal flash attention ----------------
// one wave per 16-row Q tile per (b,h). Q/K head-major [BH,T,64] f16,
// V transposed [BH,64,T] f16, output fp32 [B,T,C].
__global__ void __launch_bounds__(32, 1)
attn_kernel(const _Float16* __restrict__ Qh, const _Float16* __restrict__ Kh,
            const _Float16* __restrict__ Vt, float* __restrict__ O) {
  __shared__ _Float16 pb[16 * 32];
  int lane = threadIdx.x;
  int qb = blockIdx.x * 16;
  int bh = blockIdx.y;
  int b = bh >> 4, h = bh & 15;
  size_t bhT = (size_t)bh * TD;
  int nlo = lane & 15;
  int kA = (lane >> 4) * 8;
  int kB = (lane >> 4) * 8;
  int mrowbase = (lane >> 4) * 8;

  const _Float16* qrow = Qh + (bhT + qb + nlo) * KD + kA;
  v16h qa0 = load_fragA(qrow);        // c = 0..31
  v16h qa1 = load_fragA(qrow + 32);   // c = 32..63

  v8f o0 = {}, o1 = {}, o2 = {}, o3 = {};
  float mrow[8], lrow[8];
  #pragma unroll
  for (int v = 0; v < 8; ++v) { mrow[v] = -3.0e38f; lrow[v] = 0.f; }

  for (int kb0 = 0; kb0 < qb + 16; kb0 += 32) {
    // distinct registers per fragment: K frags + V frags issued together so
    // V loads stay in flight through the whole softmax phase.
    const _Float16* kp0 = Kh + (bhT + kb0 + nlo) * KD + kB;
    const _Float16* kp1 = Kh + (bhT + kb0 + 16 + nlo) * KD + kB;
    v16h bk00 = load_fragB(kp0);
    v16h bk01 = load_fragB(kp0 + 32);
    v16h bk10 = load_fragB(kp1);
    v16h bk11 = load_fragB(kp1 + 32);
    const _Float16* vp = Vt + ((size_t)bh * KD + nlo) * TD + kb0 + kB;
    v16h bv0 = load_fragB(vp);
    v16h bv1 = load_fragB(vp + 16 * TD);
    v16h bv2 = load_fragB(vp + 32 * TD);
    v16h bv3 = load_fragB(vp + 48 * TD);

    v8f s0 = {}, s1 = {};
    s0 = wmma16(qa0, bk00, s0);
    s0 = wmma16(qa1, bk01, s0);
    s1 = wmma16(qa0, bk10, s1);
    s1 = wmma16(qa1, bk11, s1);

    // scale + causal mask + online softmax
    float p0[8], p1[8], nm[8];
    int kc0 = kb0 + nlo, kc1 = kb0 + 16 + nlo;
    #pragma unroll
    for (int v = 0; v < 8; ++v) {
      int qr = qb + mrowbase + v;
      float a = (kc0 <= qr) ? s0[v] * 0.125f : -1e30f;
      float c = (kc1 <= qr) ? s1[v] * 0.125f : -1e30f;
      p0[v] = a; p1[v] = c;
      nm[v] = rmax16(fmaxf(a, c));
    }
    #pragma unroll
    for (int v = 0; v < 8; ++v) {
      float mnew = fmaxf(mrow[v], nm[v]);
      float corr = __expf(mrow[v] - mnew);
      mrow[v] = mnew;
      float e0 = __expf(p0[v] - mnew);
      float e1 = __expf(p1[v] - mnew);
      p0[v] = e0; p1[v] = e1;
      lrow[v] = lrow[v] * corr + rsum16(e0 + e1);
      o0[v] *= corr; o1[v] *= corr; o2[v] *= corr; o3[v] *= corr;
    }

    // stage P through LDS: C-layout -> A-layout
    __syncthreads();
    #pragma unroll
    for (int v = 0; v < 8; ++v) {
      int M = mrowbase + v;
      pb[M * 32 + nlo]      = (_Float16)p0[v];
      pb[M * 32 + 16 + nlo] = (_Float16)p1[v];
    }
    __syncthreads();
    v16h pa = load_fragA(pb + nlo * 32 + kA);

    // O += P * V
    o0 = wmma16(pa, bv0, o0);
    o1 = wmma16(pa, bv1, o1);
    o2 = wmma16(pa, bv2, o2);
    o3 = wmma16(pa, bv3, o3);
  }

  #pragma unroll
  for (int v = 0; v < 8; ++v) {
    float inv = 1.0f / lrow[v];
    int qr = qb + mrowbase + v;
    size_t base = ((size_t)b * TD + qr) * CD + h * KD + nlo;
    O[base]      = o0[v] * inv;
    O[base + 16] = o1[v] * inv;
    O[base + 32] = o2[v] * inv;
    O[base + 48] = o3[v] * inv;
  }
}

// ---------------- host-side orchestration ----------------

extern "C" void kernel_launch(void* const* d_in, const int* in_sizes, int n_in,
                              void* d_out, int out_size, void* d_ws, size_t ws_size,
                              hipStream_t stream) {
  const float* x      = (const float*)d_in[0];
  const float* sstate = (const float*)d_in[1];
  const float* tmx    = (const float*)d_in[2];
  const float* tmr    = (const float*)d_in[3];
  const float* tmk    = (const float*)d_in[4];
  const float* tmv    = (const float*)d_in[5];
  const float* w1     = (const float*)d_in[6];
  const float* w2     = (const float*)d_in[7];
  const float* Wq     = (const float*)d_in[8];
  const float* Wk     = (const float*)d_in[9];
  const float* Wv     = (const float*)d_in[10];
  const float* Wo     = (const float*)d_in[11];
  const float* ln_r_g = (const float*)d_in[12];
  const float* ln_r_b = (const float*)d_in[13];
  const float* ln_k_g = (const float*)d_in[14];
  const float* ln_k_b = (const float*)d_in[15];
  const float* ln_v_g = (const float*)d_in[16];
  const float* ln_v_b = (const float*)d_in[17];
  const float* ln_x_g = (const float*)d_in[18];
  const float* ln_x_b = (const float*)d_in[19];
  const float* ang    = (const float*)d_in[20];
  float* out = (float*)d_out;

  const size_t SZ_H = (size_t)BT * CD * sizeof(_Float16); // 8 MiB
  const size_t SZ_F = (size_t)BT * CD * sizeof(float);    // 16 MiB
  char* ws = (char*)d_ws;
  _Float16* h_xq = (_Float16*)(ws);
  _Float16* h_xk = (_Float16*)(ws + SZ_H);
  _Float16* h_xv = (_Float16*)(ws + 2 * SZ_H);
  _Float16* h_W  = (_Float16*)(ws + 3 * SZ_H);            // Wq,Wk,Wv,Wo f16 (8 MiB)
  float*    t32  = (float*)(ws + 4 * SZ_H);               // fp32 temp (16 MiB)
  _Float16* h_q  = (_Float16*)(ws + 4 * SZ_H + SZ_F);
  _Float16* h_k  = (_Float16*)(ws + 5 * SZ_H + SZ_F);
  _Float16* h_v  = (_Float16*)(ws + 6 * SZ_H + SZ_F);
  _Float16* h_o  = h_xq;                                  // reuse after GEMMs consume xq

  const int CC = CD * CD;

  // 1) mixing coefficients -> xq/xk/xv (f16)
  mix_kernel<<<BT, 256, 0, stream>>>(x, sstate, tmx, tmr, tmk, tmv, w1, w2, h_xq, h_xk, h_xv);

  // 2) weights -> f16
  cvt_f32_f16<<<(CC + 255) / 256, 256, 0, stream>>>(Wq, h_W, CC);
  cvt_f32_f16<<<(CC + 255) / 256, 256, 0, stream>>>(Wk, h_W + (size_t)CC, CC);
  cvt_f32_f16<<<(CC + 255) / 256, 256, 0, stream>>>(Wv, h_W + (size_t)2 * CC, CC);
  cvt_f32_f16<<<(CC + 255) / 256, 256, 0, stream>>>(Wo, h_W + (size_t)3 * CC, CC);

  dim3 gGemm(CD / 64, BT / 16);

  // 3) q = LN(xq@Wq^T) + rope
  gemm_nt<<<gGemm, 32, 0, stream>>>(h_xq, h_W, t32);
  ln_kernel<<<BT, 256, 0, stream>>>(t32, ln_r_g, ln_r_b, ang, h_q, 0);
  // k
  gemm_nt<<<gGemm, 32, 0, stream>>>(h_xk, h_W + (size_t)CC, t32);
  ln_kernel<<<BT, 256, 0, stream>>>(t32, ln_k_g, ln_k_b, ang, h_k, 0);
  // v (transposed per-head layout)
  gemm_nt<<<gGemm, 32, 0, stream>>>(h_xv, h_W + (size_t)2 * CC, t32);
  ln_kernel<<<BT, 256, 0, stream>>>(t32, ln_v_g, ln_v_b, ang, h_v, 1);

  // 4) causal flash attention -> t32 ([B,T,C] fp32)
  attn_kernel<<<dim3(TD / 16, BD * HD), 32, 0, stream>>>(h_q, h_k, h_v, t32);

  // 5) output LN -> f16, final projection -> d_out (fp32)
  ln_kernel<<<BT, 256, 0, stream>>>(t32, ln_x_g, ln_x_b, nullptr, h_o, 2);
  gemm_nt<<<gGemm, 32, 0, stream>>>(h_o, h_W + (size_t)3 * CC, out);

  (void)in_sizes; (void)n_in; (void)out_size; (void)ws_size;
}
